// SpaFormer_35682588295641
// MI455X (gfx1250) — matrix operand order repaired and checked by reading, not compiled, not run on hardware
//
#include <hip/hip_runtime.h>
#include <hip/hip_bf16.h>
#include <math.h>

#define BATCHB 16
#define SEQN   512
#define DFEAT  3
#define DPOS   3
#define LAYERS 4
#define HEADS  8
#define DKK    64
#define DVV    64
#define DMM    512
#define DII    2048
#define MMASK  64

typedef __attribute__((ext_vector_type(16))) __bf16 v16bf;
typedef __attribute__((ext_vector_type(8)))  float  v8f;

__device__ __forceinline__ unsigned f2bf1(float f) {
    unsigned u = __float_as_uint(f);
    return (u + 0x7FFFu + ((u >> 16) & 1u)) >> 16;
}
__device__ __forceinline__ uint2 pack4bf(float a, float b, float c, float d) {
    uint2 r;
    r.x = f2bf1(a) | (f2bf1(b) << 16);
    r.y = f2bf1(c) | (f2bf1(d) << 16);
    return r;
}

union FragBF { int4 q[2]; v16bf v; };

#define TILE_M 128
#define TILE_N 128
#define TILE_K 32
#define LDSP   40   // padded LDS pitch in bf16 elems (80B rows, 16B aligned)

// ---------------------------------------------------------------------------
// Batched-stride GEMM:  C = act(A @ B + bias)
// A (m,k) at A[(z/bdiv)*aOut + (z%bdiv)*aIn + m*lda_r + k]        (k contiguous)
// B (k,n) at B[(z/bdiv)*bOut + (z%bdiv)*bIn + k*ldb_k + n*ldb_n]
// C (m,n) at C[(z/bdiv)*cOut + (z%bdiv)*cIn + m*ldc_r + n]
// Requires K % 32 == 0.  M/N edges handled by index clamping (garbage lanes
// are computed but never stored).
// 128 threads = 4 waves; each wave owns a 64x64 subtile = 4x4 WMMA tiles.
// ---------------------------------------------------------------------------
__global__ __launch_bounds__(128)
void gemm_wmma_bf16(const float* __restrict__ A, int lda_r,
                    long long aOut, long long aIn,
                    const float* __restrict__ B, int ldb_k, int ldb_n,
                    long long bOut, long long bIn,
                    float* __restrict__ C, int ldc_r,
                    long long cOut, long long cIn,
                    const float* __restrict__ bias,
                    int M, int Ncol, int K, int act, int bdiv)
{
    __shared__ unsigned short As[TILE_M * LDSP];
    __shared__ unsigned short Bs[TILE_N * LDSP];

    const int z = blockIdx.z;
    const float* Ab = A + (long long)(z / bdiv) * aOut + (long long)(z % bdiv) * aIn;
    const float* Bb = B + (long long)(z / bdiv) * bOut + (long long)(z % bdiv) * bIn;
    float*       Cb = C + (long long)(z / bdiv) * cOut + (long long)(z % bdiv) * cIn;

    const int rowBase = blockIdx.y * TILE_M;
    const int colBase = blockIdx.x * TILE_N;

    const int tid  = threadIdx.x;     // 0..127
    const int lane = tid & 31;
    const int wave = tid >> 5;        // 0..3
    const int wm   = wave >> 1;       // 0..1 : 64-row strip
    const int wn   = wave & 1;        // 0..1 : 64-col strip
    const int lr   = lane & 15;
    const int hi   = lane >> 4;       // 0 or 1

    // B staging: one column per thread, clamped
    int gn = colBase + tid; if (gn > Ncol - 1) gn = Ncol - 1;
    const float* bcol = Bb + (long long)gn * ldb_n;
    const bool bKcontig = (ldb_k == 1);

    v8f acc[4][4];
#pragma unroll
    for (int i = 0; i < 4; i++)
#pragma unroll
        for (int j = 0; j < 4; j++)
#pragma unroll
            for (int e = 0; e < 8; e++) acc[i][j][e] = 0.0f;

    for (int k0 = 0; k0 < K; k0 += TILE_K) {
        __syncthreads();

        // ---- stage A tile [128 rows x 32 k], float4 loads -> packed b64 LDS
#pragma unroll
        for (int j = 0; j < 8; j++) {
            int e = j * 512 + tid * 4;
            int r = e >> 5, kk = e & 31;
            int gr = rowBase + r; if (gr > M - 1) gr = M - 1;
            float4 v = *(const float4*)(Ab + (long long)gr * lda_r + (k0 + kk));
            *(uint2*)&As[r * LDSP + kk] = pack4bf(v.x, v.y, v.z, v.w);
        }
        // ---- stage B tile transposed: Bs[n][k]
        if (bKcontig) {
#pragma unroll
            for (int kk = 0; kk < TILE_K; kk += 4) {
                float4 v = *(const float4*)(bcol + (k0 + kk));
                *(uint2*)&Bs[tid * LDSP + kk] = pack4bf(v.x, v.y, v.z, v.w);
            }
        } else {
#pragma unroll
            for (int kk = 0; kk < TILE_K; kk += 4) {
                float a0 = bcol[(long long)(k0 + kk + 0) * ldb_k];
                float a1 = bcol[(long long)(k0 + kk + 1) * ldb_k];
                float a2 = bcol[(long long)(k0 + kk + 2) * ldb_k];
                float a3 = bcol[(long long)(k0 + kk + 3) * ldb_k];
                *(uint2*)&Bs[tid * LDSP + kk] = pack4bf(a0, a1, a2, a3);
            }
        }
        __syncthreads();

        // speculative prefetch of next A K-tile (global_prefetch_b8)
        if (k0 + TILE_K < K) {
            int r = tid;  // one row per thread
            int gr = rowBase + r; if (gr > M - 1) gr = M - 1;
            __builtin_prefetch(Ab + (long long)gr * lda_r + (k0 + TILE_K), 0, 1);
        }

        // ---- B fragments for this wave's 4 column tiles (kept live)
        FragBF bfr[4];
#pragma unroll
        for (int nt = 0; nt < 4; nt++) {
            int c  = wn * 64 + nt * 16 + lr;
            int kb = hi ? 8 : 0;
            const int4* p = (const int4*)(&Bs[c * LDSP + kb]);
            bfr[nt].q[0] = p[0];
            bfr[nt].q[1] = p[2];
        }
        // ---- 4 row tiles x 4 col tiles of WMMA
#pragma unroll
        for (int mt = 0; mt < 4; mt++) {
            FragBF af;
            int r  = wm * 64 + mt * 16 + lr;
            int kb = hi ? 8 : 0;
            af.q[0] = *(const int4*)(&As[r * LDSP + kb]);
            af.q[1] = *(const int4*)(&As[r * LDSP + kb + 16]);
#pragma unroll
            for (int nt = 0; nt < 4; nt++)
                acc[mt][nt] = __builtin_amdgcn_wmma_f32_16x16x32_bf16(
                    false, af.v, false, bfr[nt].v,
                    (short)0, acc[mt][nt], false, false);
        }
    }

    // ---- epilogue: VGPR e holds row (hi*8+e), col = lane&15
#pragma unroll
    for (int mt = 0; mt < 4; mt++)
#pragma unroll
        for (int nt = 0; nt < 4; nt++) {
            int colg = colBase + wn * 64 + nt * 16 + lr;
            if (colg >= Ncol) continue;
            float bval = bias ? bias[colg] : 0.0f;
#pragma unroll
            for (int e = 0; e < 8; e++) {
                int rowg = rowBase + wm * 64 + mt * 16 + hi * 8 + e;
                if (rowg >= M) continue;
                float v = acc[mt][nt][e] + bval;
                if (act == 1)      v = fmaxf(v, 0.0f);
                else if (act == 2) v = 0.5f * v * (1.0f + erff(v * 0.7071067811865475f));
                Cb[(long long)rowg * ldc_r + colg] = v;
            }
        }
}

// ---------------------------------------------------------------------------
// feature encoder first linear: out[i,j] = sum_p feat[i,p]*w1[p,j] + b1[j]
// ---------------------------------------------------------------------------
__global__ void enc1_kernel(const float* __restrict__ feat, const float* __restrict__ w1,
                            const float* __restrict__ b1, float* __restrict__ out, int rows)
{
    long long t = (long long)blockIdx.x * blockDim.x + threadIdx.x;
    long long total = (long long)rows * DMM;
    if (t >= total) return;
    int i = (int)(t / DMM), j = (int)(t % DMM);
    float s = b1[j];
#pragma unroll
    for (int p = 0; p < DFEAT; p++) s += feat[i * DFEAT + p] * w1[p * DMM + j];
    out[t] = s;
}

// ---------------------------------------------------------------------------
// LayerNorm (row length 512), optional fused residual: y = LN(in + res)*g + b
// ---------------------------------------------------------------------------
__global__ __launch_bounds__(128)
void ln_kernel(const float* __restrict__ in, const float* __restrict__ res,
               const float* __restrict__ g, const float* __restrict__ bt,
               float* __restrict__ out)
{
    const int row = blockIdx.x;
    const int tid = threadIdx.x;
    const float* ip = in + (long long)row * DMM;
    const float* rp = res ? res + (long long)row * DMM : nullptr;
    float v[4];
    float s = 0.0f;
#pragma unroll
    for (int i = 0; i < 4; i++) {
        int j = tid + 128 * i;
        float x = ip[j];
        if (rp) x += rp[j];
        v[i] = x; s += x;
    }
    __shared__ float sb[128];
    sb[tid] = s; __syncthreads();
    for (int off = 64; off > 0; off >>= 1) {
        if (tid < off) sb[tid] += sb[tid + off];
        __syncthreads();
    }
    float mu = sb[0] * (1.0f / DMM);
    __syncthreads();
    float s2 = 0.0f;
#pragma unroll
    for (int i = 0; i < 4; i++) { float d = v[i] - mu; s2 += d * d; }
    sb[tid] = s2; __syncthreads();
    for (int off = 64; off > 0; off >>= 1) {
        if (tid < off) sb[tid] += sb[tid + off];
        __syncthreads();
    }
    float inv = rsqrtf(sb[0] * (1.0f / DMM) + 1e-6f);
#pragma unroll
    for (int i = 0; i < 4; i++) {
        int j = tid + 128 * i;
        out[(long long)row * DMM + j] = (v[i] - mu) * inv * g[j] + bt[j];
    }
}

// ---------------------------------------------------------------------------
// qw[b,n,h,p] = sum_k q[b,n,h,k]*wp[p,h,k] ;  qb[b,n,h] = sum_k q*bp[h,k]
// ---------------------------------------------------------------------------
__global__ void qwqb_kernel(const float* __restrict__ q, const float* __restrict__ wp,
                            const float* __restrict__ bp,
                            float* __restrict__ qw, float* __restrict__ qb)
{
    int t = blockIdx.x * blockDim.x + threadIdx.x;
    int bnh = t >> 2, which = t & 3;
    if (bnh >= BATCHB * SEQN * HEADS) return;
    int h = bnh % HEADS;
    const float* qp  = q + (long long)bnh * DKK;
    const float* wpt = (which < 3) ? (wp + (which * HEADS + h) * DKK) : (bp + h * DKK);
    float s = 0.0f;
    for (int k = 0; k < DKK; k++) s += qp[k] * wpt[k];
    if (which < 3) qw[(long long)bnh * 3 + which] = s;
    else           qb[bnh] = s;
}

// ---------------------------------------------------------------------------
// fused relative-position bias + scale + softmax over one score row
// blockIdx.x = n, blockIdx.y = z = b*H + h ; 256 threads, 2 cols each
// ---------------------------------------------------------------------------
__global__ __launch_bounds__(256)
void softmax_rel_kernel(float* __restrict__ sc, const float* __restrict__ rpos,
                        const float* __restrict__ qw, const float* __restrict__ qb,
                        float scale)
{
    const int n = blockIdx.x;
    const int z = blockIdx.y;
    const int b = z / HEADS;
    const int h = z % HEADS;
    const long long bn = (long long)b * SEQN + n;
    const float* qwp = qw + (bn * HEADS + h) * 3;
    const float  qb0 = qb[bn * HEADS + h];
    const float w0 = qwp[0], w1 = qwp[1], w2 = qwp[2];
    float* row = sc + ((long long)z * SEQN + n) * SEQN;
    const float* rrow = rpos + bn * SEQN * 3;
    const int tid = threadIdx.x;

    float vals[2];
    float mx = -1e30f;
#pragma unroll
    for (int i = 0; i < 2; i++) {
        int m = tid + 256 * i;
        const float* rp = rrow + (long long)m * 3;
        float vv = (row[m] + w0 * rp[0] + w1 * rp[1] + w2 * rp[2] + qb0) * scale;
        vals[i] = vv; mx = fmaxf(mx, vv);
    }
    __shared__ float sb[256];
    sb[tid] = mx; __syncthreads();
    for (int off = 128; off > 0; off >>= 1) {
        if (tid < off) sb[tid] = fmaxf(sb[tid], sb[tid + off]);
        __syncthreads();
    }
    mx = sb[0]; __syncthreads();
    float s = 0.0f;
#pragma unroll
    for (int i = 0; i < 2; i++) { vals[i] = __expf(vals[i] - mx); s += vals[i]; }
    sb[tid] = s; __syncthreads();
    for (int off = 128; off > 0; off >>= 1) {
        if (tid < off) sb[tid] += sb[tid + off];
        __syncthreads();
    }
    float inv = 1.0f / sb[0];
#pragma unroll
    for (int i = 0; i < 2; i++) row[tid + 256 * i] = vals[i] * inv;
}

// ---------------------------------------------------------------------------
// gather masked positions: h1[(b*M+m), :] = x[b, masked_pos[b,m], :]
// ---------------------------------------------------------------------------
__global__ void gather_kernel(const float* __restrict__ X, const int* __restrict__ mpos,
                              float* __restrict__ H1)
{
    int t = blockIdx.x * blockDim.x + threadIdx.x;
    if (t >= BATCHB * MMASK * DMM) return;
    int j = t & (DMM - 1);
    int r = t >> 9;
    int b = r / MMASK;
    int pos = mpos[r];
    H1[t] = X[((long long)b * SEQN + pos) * DMM + j];
}

// ---------------------------------------------------------------------------
// final decoder column: dec[i] = h[i,:] . w2 + b2
// ---------------------------------------------------------------------------
__global__ void dec2_kernel(const float* __restrict__ h, const float* __restrict__ w2,
                            const float* __restrict__ b2, float* __restrict__ dec)
{
    int t = blockIdx.x * blockDim.x + threadIdx.x;
    if (t >= BATCHB * MMASK) return;
    float s = b2[0];
    const float* hp = h + (long long)t * DMM;
    for (int j = 0; j < DMM; j++) s += hp[j] * w2[j];
    dec[t] = s;
}

// ===========================================================================
extern "C" void kernel_launch(void* const* d_in, const int* in_sizes, int n_in,
                              void* d_out, int out_size, void* d_ws, size_t ws_size,
                              hipStream_t stream)
{
    const float* feat   = (const float*)d_in[0];
    const float* rpos   = (const float*)d_in[1];
    const int*   mpos   = (const int*)  d_in[2];
    const float* fe_w1  = (const float*)d_in[3];
    const float* fe_b1  = (const float*)d_in[4];
    const float* fe_w2  = (const float*)d_in[5];
    const float* fe_b2  = (const float*)d_in[6];
    const float* wq     = (const float*)d_in[7];
    const float* bq     = (const float*)d_in[8];
    const float* wk     = (const float*)d_in[9];
    const float* bk     = (const float*)d_in[10];
    const float* wv     = (const float*)d_in[11];
    const float* bv     = (const float*)d_in[12];
    const float* wp     = (const float*)d_in[13];
    const float* bp     = (const float*)d_in[14];
    const float* wo     = (const float*)d_in[15];
    const float* bo     = (const float*)d_in[16];
    const float* ln1_g  = (const float*)d_in[17];
    const float* ln1_b  = (const float*)d_in[18];
    const float* ffn_w1 = (const float*)d_in[19];
    const float* ffn_b1 = (const float*)d_in[20];
    const float* ffn_w2 = (const float*)d_in[21];
    const float* ffn_b2 = (const float*)d_in[22];
    const float* ln2_g  = (const float*)d_in[23];
    const float* ln2_b  = (const float*)d_in[24];
    const float* ln_g   = (const float*)d_in[25];
    const float* ln_b   = (const float*)d_in[26];
    const float* lin_w  = (const float*)d_in[27];
    const float* lin_b  = (const float*)d_in[28];
    const float* dec_w1 = (const float*)d_in[29];
    const float* dec_b1 = (const float*)d_in[30];
    const float* dec_w2 = (const float*)d_in[31];
    const float* dec_b2 = (const float*)d_in[32];

    float* out = (float*)d_out;
    float* ws  = (float*)d_ws;

    const long long ROWS = (long long)BATCHB * SEQN;    // 8192
    float* X   = ws;                                    // 8192*512
    float* TMP = X   + ROWS * DMM;                      // 8192*2048
    float* Qb  = TMP + ROWS * DII;                      // 8192*512
    float* Kb  = Qb  + ROWS * DMM;
    float* Vb  = Kb  + ROWS * DMM;
    float* SC  = Vb  + ROWS * DMM;                      // 128*512*512
    float* AO  = SC  + (long long)BATCHB * HEADS * SEQN * SEQN;
    float* QWs = AO  + ROWS * DMM;                      // 8192*8*3
    float* QBs = QWs + ROWS * HEADS * 3;                // 8192*8

    float* DEC = out;                                             // 1024
    float* H1  = out + (long long)BATCHB * MMASK;                 // 16*64*512
    float* H2  = H1  + (long long)BATCHB * MMASK * DMM;

    auto gemm = [&](const float* A, int lda_r, long long aOut, long long aIn,
                    const float* B, int ldb_k, int ldb_n, long long bOut, long long bIn,
                    float* C, int ldc_r, long long cOut, long long cIn,
                    const float* bias, int M, int N, int K, int act, int batches, int bdiv) {
        dim3 grid((N + TILE_N - 1) / TILE_N, (M + TILE_M - 1) / TILE_M, batches);
        gemm_wmma_bf16<<<grid, 128, 0, stream>>>(A, lda_r, aOut, aIn,
                                                 B, ldb_k, ldb_n, bOut, bIn,
                                                 C, ldc_r, cOut, cIn,
                                                 bias, M, N, K, act, bdiv);
    };
    auto lin = [&](const float* A, const float* W, const float* bias, float* C,
                   int M, int N, int K, int act) {
        gemm(A, K, 0, 0, W, N, 1, 0, 0, C, N, 0, 0, bias, M, N, K, act, 1, 1);
    };

    // ---- feature encoder + shared LN ----
    enc1_kernel<<<(int)((ROWS * DMM + 255) / 256), 256, 0, stream>>>(feat, fe_w1, fe_b1, TMP, (int)ROWS);
    lin(TMP, fe_w2, fe_b2, X, (int)ROWS, DMM, DMM, 0);
    ln_kernel<<<(int)ROWS, 128, 0, stream>>>(X, nullptr, ln_g, ln_b, X);

    const long long QKV_OUT = (long long)SEQN * HEADS * DKK;    // 262144
    const long long SC_OUT  = (long long)HEADS * SEQN * SEQN;   // 2097152
    const long long SC_IN   = (long long)SEQN * SEQN;           // 262144

    for (int l = 0; l < LAYERS; l++) {
        // ---- QKV projections ----
        lin(X, wq + (long long)l * DMM * DMM, bq + l * DMM, Qb, (int)ROWS, DMM, DMM, 0);
        lin(X, wk + (long long)l * DMM * DMM, bk + l * DMM, Kb, (int)ROWS, DMM, DMM, 0);
        lin(X, wv + (long long)l * DMM * DMM, bv + l * DMM, Vb, (int)ROWS, DMM, DMM, 0);

        // ---- relative-position projections ----
        qwqb_kernel<<<(BATCHB * SEQN * HEADS * 4 + 255) / 256, 256, 0, stream>>>(
            Qb, wp + (long long)l * DPOS * HEADS * DKK, bp + (long long)l * HEADS * DKK, QWs, QBs);

        // ---- scores = Q @ K^T, batched over z = b*H + h (B read k-contiguous) ----
        gemm(Qb, HEADS * DKK, QKV_OUT, DKK,
             Kb, 1, HEADS * DKK, QKV_OUT, DKK,
             SC, SEQN, SC_OUT, SC_IN,
             nullptr, SEQN, SEQN, DKK, 0, BATCHB * HEADS, HEADS);

        // ---- rel bias + scale + softmax (in place) ----
        softmax_rel_kernel<<<dim3(SEQN, BATCHB * HEADS), 256, 0, stream>>>(
            SC, rpos, QWs, QBs, 0.125f);

        // ---- out = attn @ V ----
        gemm(SC, SEQN, SC_OUT, SC_IN,
             Vb, HEADS * DVV, 1, QKV_OUT, DVV,
             AO, HEADS * DVV, QKV_OUT, DVV,
             nullptr, SEQN, DVV, SEQN, 0, BATCHB * HEADS, HEADS);

        // ---- output projection + residual LN ----
        lin(AO, wo + (long long)l * DMM * DMM, bo + l * DMM, TMP, (int)ROWS, DMM, DMM, 0);
        ln_kernel<<<(int)ROWS, 128, 0, stream>>>(TMP, X, ln1_g + l * DMM, ln1_b + l * DMM, X);

        // ---- FFN ----
        lin(X, ffn_w1 + (long long)l * DMM * DII, ffn_b1 + l * DII, TMP, (int)ROWS, DII, DMM, 1);
        lin(TMP, ffn_w2 + (long long)l * DII * DMM, ffn_b2 + l * DMM, AO, (int)ROWS, DMM, DII, 0);
        ln_kernel<<<(int)ROWS, 128, 0, stream>>>(AO, X, ln2_g + l * DMM, ln2_b + l * DMM, X);
    }

    // ---- masked gather + head ----
    const int MR = BATCHB * MMASK;   // 1024
    gather_kernel<<<(MR * DMM + 255) / 256, 256, 0, stream>>>(X, mpos, H1);
    lin(H1, lin_w, lin_b, TMP, MR, DMM, DMM, 2);                 // erf-gelu
    ln_kernel<<<MR, 128, 0, stream>>>(TMP, nullptr, ln_g, ln_b, H2);
    lin(H2, dec_w1, dec_b1, TMP, MR, DMM, DMM, 0);
    dec2_kernel<<<(MR + 255) / 256, 256, 0, stream>>>(TMP, dec_w2, dec_b2, DEC);
}
